// GATSegmentationModel_35914516529214
// MI455X (gfx1250) — matrix operand order repaired and checked by reading, not compiled, not run on hardware
//
#include <hip/hip_runtime.h>
#include <hip/hip_bf16.h>

// ---- CDNA5 vector types for WMMA -------------------------------------------
typedef __attribute__((ext_vector_type(16))) _Float16 v16h;
typedef __attribute__((ext_vector_type(8)))  _Float16 v8h;
typedef __attribute__((ext_vector_type(8)))  float    v8f;

#define NEG_SLOPE 0.2f

// Order-preserving float <-> uint map so atomicMax works for any-sign floats.
__device__ __forceinline__ unsigned fenc(float x) {
  unsigned u = __float_as_uint(x);
  return (u & 0x80000000u) ? ~u : (u | 0x80000000u);
}
__device__ __forceinline__ float fdec(unsigned u) {
  u = (u & 0x80000000u) ? (u & 0x7FFFFFFFu) : ~u;
  return __uint_as_float(u);
}
#define ENC_NEG_INF 0x007FFFFFu   // fenc(-inf)

// ---- K0: init scratch -------------------------------------------------------
__global__ void k_init(unsigned* m1, float* z1, float* g1,
                       unsigned* m2, float* z2, float* num2, int N) {
  int i = blockIdx.x * blockDim.x + threadIdx.x;
  if (i < N * 4) { m1[i] = ENC_NEG_INF; z1[i] = 0.f; g1[i] = 0.f; }
  if (i < N)     { m2[i] = ENC_NEG_INF; z2[i] = 0.f; num2[i] = 0.f; }
}

// ---- K0b: tiny precompute: s1[h], d1[h]; W2 -> fp16 -------------------------
__global__ void k_pre(const float* __restrict__ W1, const float* __restrict__ as1,
                      const float* __restrict__ ad1, const float* __restrict__ W2,
                      float* s1, float* d1, _Float16* w2h) {
  int t = threadIdx.x;           // one block of 256
  w2h[t] = (_Float16)W2[t];
  if (t < 4) {
    float ss = 0.f, dd = 0.f;
    for (int c = 0; c < 64; ++c) {
      float w = W1[t * 64 + c];
      ss += w * as1[t * 64 + c];
      dd += w * ad1[t * 64 + c];
    }
    s1[t] = ss; d1[t] = dd;
  }
}

// ---- K1: conv1 segment-max of leaky_relu(x_s*s1 + x_d*d1) over dst ----------
__global__ void k1_max(const float* __restrict__ x, const int* __restrict__ ei,
                       int E, int N, const float* __restrict__ s1,
                       const float* __restrict__ d1, unsigned* m1) {
  int e = blockIdx.x * blockDim.x + threadIdx.x;
  int Et = E + N;
  if (e >= Et) return;
  int s, d;
  if (e < E) { s = ei[e]; d = ei[E + e]; } else { s = d = e - E; }
  float xs = x[s], xd = x[d];
#pragma unroll
  for (int h = 0; h < 4; ++h) {
    float v = xs * s1[h] + xd * d1[h];
    v = v > 0.f ? v : NEG_SLOPE * v;
    atomicMax(m1 + d * 4 + h, fenc(v));
  }
}

// ---- K2: conv1 exp-sum + weighted numerator ---------------------------------
__global__ void k2_sum(const float* __restrict__ x, const int* __restrict__ ei,
                       int E, int N, const float* __restrict__ s1,
                       const float* __restrict__ d1, const unsigned* __restrict__ m1,
                       float* z1, float* g1) {
  int e = blockIdx.x * blockDim.x + threadIdx.x;
  int Et = E + N;
  if (e >= Et) return;
  if (e + 2048 < E) __builtin_prefetch(ei + e + 2048, 0, 1);  // global_prefetch_b8
  int s, d;
  if (e < E) { s = ei[e]; d = ei[E + e]; } else { s = d = e - E; }
  float xs = x[s], xd = x[d];
#pragma unroll
  for (int h = 0; h < 4; ++h) {
    float v = xs * s1[h] + xd * d1[h];
    v = v > 0.f ? v : NEG_SLOPE * v;
    float t = __expf(v - fdec(m1[d * 4 + h]));
    atomicAdd(z1 + d * 4 + h, t);
    atomicAdd(g1 + d * 4 + h, t * xs);
  }
}

// ---- K3: materialize h1 = relu(g*W1 + b1) as fp16 [N,256] -------------------
__global__ void k3_h1(const float* __restrict__ g1, const float* __restrict__ z1,
                      const float* __restrict__ W1, const float* __restrict__ b1,
                      _Float16* __restrict__ h1, int N) {
  int idx = blockIdx.x * blockDim.x + threadIdx.x;
  if (idx >= N * 256) return;
  int n = idx >> 8, j = idx & 255, h = j >> 6;
  float g = g1[n * 4 + h] / (z1[n * 4 + h] + 1e-16f);
  float v = g * W1[j] + b1[j];
  h1[idx] = (_Float16)(v > 0.f ? v : 0.f);
}

// ---- K4: hW2 = h1[N,256] @ W2[256] via v_wmma_f32_16x16x32_f16 --------------
// One wave handles a 16-row tile, K=256 -> 8 chained WMMAs.
// A fragment: lane half 'hi' holds K = hi*16..hi*16+15 contiguously (same order
// as B). B fragment: lane half holds
// K = hi*16..hi*16+15 contiguously; W2 replicated over all 16 columns.
__global__ void k4_gemv(const _Float16* __restrict__ h1,
                        const _Float16* __restrict__ w2h,
                        float* __restrict__ hW2, int ntiles) {
  int wave = (int)((blockIdx.x * blockDim.x + threadIdx.x) >> 5);
  int lane = threadIdx.x & 31;
  if (wave >= ntiles) return;                 // wave-uniform guard: EXEC stays all-1
  int hi = lane >> 4;
  const _Float16* arow = h1 + (size_t)((wave << 4) + (lane & 15)) * 256;
  v8f c = {};
#pragma unroll
  for (int kb = 0; kb < 8; ++kb) {
    int k0 = kb * 32;
    v16h a = *(const v16h*)(arow + k0 + hi * 16);       // K contiguous per half
    v16h b = *(const v16h*)(w2h + k0 + hi * 16);        // K contiguous per half
    c = __builtin_amdgcn_wmma_f32_16x16x32_f16(false, a, false, b,
                                               (short)0, c, false, false);
  }
  // D layout: VGPR v, lanes 0-15 -> M=v, lanes 16-31 -> M=v+8; N = lane&15.
  if ((lane & 15) == 0) {
    int base = (wave << 4) + hi * 8;
#pragma unroll
    for (int v = 0; v < 8; ++v) hW2[base + v] = c[v];
  }
}

// ---- K5: conv2 segment-max --------------------------------------------------
__global__ void k5_max(const float* __restrict__ hW2, const int* __restrict__ ei,
                       int E, int N, const float* __restrict__ a2s_p,
                       const float* __restrict__ a2d_p, unsigned* m2) {
  int e = blockIdx.x * blockDim.x + threadIdx.x;
  int Et = E + N;
  if (e >= Et) return;
  int s, d;
  if (e < E) { s = ei[e]; d = ei[E + e]; } else { s = d = e - E; }
  float v = hW2[s] * a2s_p[0] + hW2[d] * a2d_p[0];
  v = v > 0.f ? v : NEG_SLOPE * v;
  atomicMax(m2 + d, fenc(v));
}

// ---- K6: conv2 exp-sum + weighted numerator ---------------------------------
__global__ void k6_sum(const float* __restrict__ hW2, const int* __restrict__ ei,
                       int E, int N, const float* __restrict__ a2s_p,
                       const float* __restrict__ a2d_p, const unsigned* __restrict__ m2,
                       float* z2, float* num2) {
  int e = blockIdx.x * blockDim.x + threadIdx.x;
  int Et = E + N;
  if (e >= Et) return;
  if (e + 2048 < E) __builtin_prefetch(ei + e + 2048, 0, 1);
  int s, d;
  if (e < E) { s = ei[e]; d = ei[E + e]; } else { s = d = e - E; }
  float hs = hW2[s];
  float v = hs * a2s_p[0] + hW2[d] * a2d_p[0];
  v = v > 0.f ? v : NEG_SLOPE * v;
  float t = __expf(v - fdec(m2[d]));
  atomicAdd(z2 + d, t);
  atomicAdd(num2 + d, t * hs);
}

// ---- K7: finalize -----------------------------------------------------------
__global__ void k7_out(const float* __restrict__ num2, const float* __restrict__ z2,
                       const float* __restrict__ b2, float* __restrict__ out, int N) {
  int n = blockIdx.x * blockDim.x + threadIdx.x;
  if (n >= N) return;
  out[n] = num2[n] / (z2[n] + 1e-16f) + b2[0];
}

// ---- launcher ---------------------------------------------------------------
extern "C" void kernel_launch(void* const* d_in, const int* in_sizes, int n_in,
                              void* d_out, int out_size, void* d_ws, size_t ws_size,
                              hipStream_t stream) {
  const float* x   = (const float*)d_in[0];
  const int*   ei  = (const int*)d_in[1];       // [2,E] int32
  const float* W1  = (const float*)d_in[2];
  const float* as1 = (const float*)d_in[3];
  const float* ad1 = (const float*)d_in[4];
  const float* b1  = (const float*)d_in[5];
  const float* W2  = (const float*)d_in[6];
  const float* as2 = (const float*)d_in[7];
  const float* ad2 = (const float*)d_in[8];
  const float* b2  = (const float*)d_in[9];
  float* out = (float*)d_out;

  const int N = in_sizes[0];            // IN_DIM == 1
  const int E = in_sizes[1] / 2;
  const size_t sN = (size_t)N;
  char* ws = (char*)d_ws;

  float*    s1   = (float*)(ws + 0);
  float*    d1   = (float*)(ws + 16);
  _Float16* w2h  = (_Float16*)(ws + 64);                 // 512 B
  unsigned* m1   = (unsigned*)(ws + 1024);               // 16N
  float*    z1   = (float*)(ws + 1024 + 16 * sN);        // 16N
  float*    g1   = (float*)(ws + 1024 + 32 * sN);        // 16N
  unsigned* m2   = (unsigned*)(ws + 1024 + 48 * sN);     // 4N
  float*    z2   = (float*)(ws + 1024 + 52 * sN);        // 4N
  float*    num2 = (float*)(ws + 1024 + 56 * sN);        // 4N
  float*    hW2  = (float*)(ws + 1024 + 60 * sN);        // 4N
  _Float16* h1   = (_Float16*)(ws + 1024 + 64 * sN);     // 512N

  const int TB = 256;
  const int Et = E + N;
  const int gE = (Et + TB - 1) / TB;

  k_init<<<(N * 4 + TB - 1) / TB, TB, 0, stream>>>(m1, z1, g1, m2, z2, num2, N);
  k_pre<<<1, 256, 0, stream>>>(W1, as1, ad1, W2, s1, d1, w2h);

  k1_max<<<gE, TB, 0, stream>>>(x, ei, E, N, s1, d1, m1);
  k2_sum<<<gE, TB, 0, stream>>>(x, ei, E, N, s1, d1, m1, z1, g1);
  k3_h1<<<(N * 256 + TB - 1) / TB, TB, 0, stream>>>(g1, z1, W1, b1, h1, N);

  int ntiles = N / 16;                                   // N = 65536 -> 4096
  int wavesPerBlock = TB / 32;                           // 8
  k4_gemv<<<(ntiles + wavesPerBlock - 1) / wavesPerBlock, TB, 0, stream>>>(
      h1, w2h, hW2, ntiles);

  k5_max<<<gE, TB, 0, stream>>>(hW2, ei, E, N, as2, ad2, m2);
  k6_sum<<<gE, TB, 0, stream>>>(hW2, ei, E, N, as2, ad2, m2, z2, num2);
  k7_out<<<(N + TB - 1) / TB, TB, 0, stream>>>(num2, z2, b2, out, N);
}
